// MixtralMoE_2087354105877
// MI455X (gfx1250) — compile-verified, matrix operand order checked by reading, not run on hardware
//
#include <hip/hip_runtime.h>

#define T_TOK 8192
#define E_EXP 8
#define H_DIM 2048
#define I_DIM 7168

typedef __bf16 bf16x16 __attribute__((ext_vector_type(16)));
typedef float v8f __attribute__((ext_vector_type(8)));

__device__ inline __bf16 f2bf(float f) { return (__bf16)f; }

// ---- gfx1250 async global<->LDS (ASYNCcnt) helpers --------------------------
__device__ inline void async_ld16(void* lds, const void* g) {
    asm volatile("global_load_async_to_lds_b128 %0, %1, off"
                 :: "v"((unsigned)(size_t)lds), "v"(g) : "memory");
}
__device__ inline void async_st16(void* g, const void* lds) {
    asm volatile("global_store_async_from_lds_b128 %0, %1, off"
                 :: "v"(g), "v"((unsigned)(size_t)lds) : "memory");
}
__device__ inline void wait_async0() {
    asm volatile("s_wait_asynccnt 0x0" ::: "memory");
}

// ---------------------------------------------------------------- init
__global__ void moe_zero(int* counts, float* zsrc) {
    const int tid = threadIdx.x;
    if (tid < E_EXP) counts[tid] = 0;
    if (tid >= 8 && tid < 24) zsrc[tid - 8] = 0.f;  // 64B zero source
}

// ---------------------------------------------------------------- router
__global__ __launch_bounds__(256) void moe_router(
    const float* __restrict__ x, const float* __restrict__ gate,
    float* __restrict__ routew, int* __restrict__ counts, int* __restrict__ lists) {
    const int wid  = threadIdx.x >> 5;
    const int lane = threadIdx.x & 31;
    const int t    = blockIdx.x * 8 + wid;

    float acc[E_EXP];
#pragma unroll
    for (int e = 0; e < E_EXP; ++e) acc[e] = 0.f;

    const float* xr = x + (size_t)t * H_DIM;
    for (int h = lane * 4; h < H_DIM; h += 32 * 4) {
        float4 xv = *(const float4*)(xr + h);
#pragma unroll
        for (int e = 0; e < E_EXP; ++e) {
            float4 gv = *(const float4*)(gate + e * H_DIM + h);
            acc[e] += xv.x * gv.x + xv.y * gv.y + xv.z * gv.z + xv.w * gv.w;
        }
    }
#pragma unroll
    for (int e = 0; e < E_EXP; ++e)
        for (int off = 16; off; off >>= 1) acc[e] += __shfl_xor(acc[e], off, 32);

    if (lane == 0) {
        int i0 = 0; float b0 = acc[0];
#pragma unroll
        for (int e = 1; e < E_EXP; ++e) if (acc[e] > b0) { b0 = acc[e]; i0 = e; }
        int i1 = (i0 == 0) ? 1 : 0; float b1 = acc[i1];
#pragma unroll
        for (int e = 0; e < E_EXP; ++e)
            if (e != i0 && acc[e] > b1) { b1 = acc[e]; i1 = e; }
        float e1  = __expf(b1 - b0);
        float inv = 1.f / (1.f + e1);
        routew[t * 2 + 0] = inv;
        routew[t * 2 + 1] = e1 * inv;
        int p0 = atomicAdd(&counts[i0], 1);
        lists[i0 * T_TOK + p0] = (t << 1);
        int p1 = atomicAdd(&counts[i1], 1);
        lists[i1 * T_TOK + p1] = (t << 1) | 1;
    }
}

// ---------------------------------------------------------------- GEMM1 + SwiGLU
struct FFN1W { __bf16 b1[128][32]; __bf16 b3[128][32]; };
union  FFN1U { FFN1W w; __bf16 outT[128][128]; };

__global__ __launch_bounds__(256) void moe_ffn1(
    const float* __restrict__ x, const float* __restrict__ w1,
    const float* __restrict__ w3, const int* __restrict__ counts,
    const int* __restrict__ lists, __bf16* __restrict__ hmid,
    __bf16* __restrict__ dump) {
    const int e   = blockIdx.z;
    const int cnt = counts[e];
    const int m0  = blockIdx.y * 128;
    if (m0 >= cnt) return;
    const int n0  = blockIdx.x * 128;

    __shared__ __align__(64) __bf16 sA[128][32];
    __shared__ __align__(64) FFN1U  sU;
    __shared__ int sAid[128];

    const int tid  = threadIdx.x;
    const int lane = tid & 31;
    const int wid  = tid >> 5;
    const int wm   = (wid >> 2) * 64;
    const int wn   = (wid & 3) * 32;

    const float* w1e   = w1 + (size_t)e * I_DIM * H_DIM;
    const float* w3e   = w3 + (size_t)e * I_DIM * H_DIM;
    const int*   listE = lists + e * T_TOK + m0;

    v8f cg[4][2], cu[4][2];
#pragma unroll
    for (int mi = 0; mi < 4; ++mi)
#pragma unroll
        for (int ni = 0; ni < 2; ++ni) {
            cg[mi][ni] = (v8f){0, 0, 0, 0, 0, 0, 0, 0};
            cu[mi][ni] = (v8f){0, 0, 0, 0, 0, 0, 0, 0};
        }

    const int r  = tid >> 3;
    const int c4 = (tid & 7) * 4;

    // software pipeline: next k-tile held in VGPRs while WMMAs run
    float4 pA[4], pB1[4], pB3[4];
    int    pAid[4];
    auto issue = [&](int k0) {
#pragma unroll
        for (int it = 0; it < 4; ++it) {
            const int row = it * 32 + r;
            pAid[it] = -1;
            pA[it]   = make_float4(0.f, 0.f, 0.f, 0.f);
            if (m0 + row < cnt) {
                const int entry = listE[row];
                pAid[it] = entry;
                pA[it] = *(const float4*)(x + (size_t)(entry >> 1) * H_DIM + k0 + c4);
            }
            const size_t wb = (size_t)(n0 + row) * H_DIM + k0 + c4;
            pB1[it] = *(const float4*)(w1e + wb);
            pB3[it] = *(const float4*)(w3e + wb);
            __builtin_prefetch(w1e + wb + 64, 0, 1);
            __builtin_prefetch(w3e + wb + 64, 0, 1);
        }
    };
    issue(0);

    for (int k0 = 0; k0 < H_DIM; k0 += 32) {
        __syncthreads();   // previous step's fragment reads complete
#pragma unroll
        for (int it = 0; it < 4; ++it) {
            const int row = it * 32 + r;
            sA[row][c4 + 0] = f2bf(pA[it].x);
            sA[row][c4 + 1] = f2bf(pA[it].y);
            sA[row][c4 + 2] = f2bf(pA[it].z);
            sA[row][c4 + 3] = f2bf(pA[it].w);
            if ((tid & 7) == 0) sAid[row] = pAid[it];
            sU.w.b1[row][c4 + 0] = f2bf(pB1[it].x); sU.w.b1[row][c4 + 1] = f2bf(pB1[it].y);
            sU.w.b1[row][c4 + 2] = f2bf(pB1[it].z); sU.w.b1[row][c4 + 3] = f2bf(pB1[it].w);
            sU.w.b3[row][c4 + 0] = f2bf(pB3[it].x); sU.w.b3[row][c4 + 1] = f2bf(pB3[it].y);
            sU.w.b3[row][c4 + 2] = f2bf(pB3[it].z); sU.w.b3[row][c4 + 3] = f2bf(pB3[it].w);
        }
        __syncthreads();
        if (k0 + 32 < H_DIM) issue(k0 + 32);   // overlap HBM latency with WMMA

        const int khalf = (lane >> 4) * 16;
        const int mrow  = lane & 15;
        bf16x16 af[4], b1f[2], b3f[2];
#pragma unroll
        for (int mi = 0; mi < 4; ++mi)
            af[mi] = *(const bf16x16*)&sA[wm + mi * 16 + mrow][khalf];
#pragma unroll
        for (int ni = 0; ni < 2; ++ni) {
            b1f[ni] = *(const bf16x16*)&sU.w.b1[wn + ni * 16 + mrow][khalf];
            b3f[ni] = *(const bf16x16*)&sU.w.b3[wn + ni * 16 + mrow][khalf];
        }
#pragma unroll
        for (int mi = 0; mi < 4; ++mi)
#pragma unroll
            for (int ni = 0; ni < 2; ++ni) {
                cg[mi][ni] = __builtin_amdgcn_wmma_f32_16x16x32_bf16(
                    false, af[mi], false, b1f[ni], (short)0, cg[mi][ni], false, false);
                cu[mi][ni] = __builtin_amdgcn_wmma_f32_16x16x32_bf16(
                    false, af[mi], false, b3f[ni], (short)0, cu[mi][ni], false, false);
            }
    }

    // epilogue: SwiGLU into LDS tile, then coalesced async stores from LDS
    __syncthreads();
    const int nlane = lane & 15;
    const int mbase = (lane >> 4) * 8;
#pragma unroll
    for (int mi = 0; mi < 4; ++mi)
#pragma unroll
        for (int ni = 0; ni < 2; ++ni) {
            const int col = wn + ni * 16 + nlane;
#pragma unroll
            for (int j = 0; j < 8; ++j) {
                const int mloc = wm + mi * 16 + mbase + j;
                float g = cg[mi][ni][j];
                float u = cu[mi][ni][j];
                sU.outT[mloc][col] = f2bf((g / (1.f + __expf(-g))) * u);
            }
        }
    __syncthreads();
#pragma unroll
    for (int it = 0; it < 8; ++it) {
        const int u   = it * 256 + tid;
        const int row = u >> 4;
        const int seg = u & 15;
        const int aid = sAid[row];
        __bf16* g = (aid >= 0) ? (hmid + (size_t)aid * I_DIM + n0 + seg * 8) : dump;
        async_st16(g, &sU.outT[row][seg * 8]);
    }
    wait_async0();
}

// ---------------------------------------------------------------- GEMM2
union FFN2U { __bf16 b[128][32]; float outF[128][128]; };

__global__ __launch_bounds__(256) void moe_ffn2(
    const __bf16* __restrict__ hmid, const float* __restrict__ w2,
    const int* __restrict__ counts, const int* __restrict__ lists,
    float* __restrict__ acc, const __bf16* __restrict__ zsrc,
    float* __restrict__ dump) {
    const int e   = blockIdx.z;
    const int cnt = counts[e];
    const int m0  = blockIdx.y * 128;
    if (m0 >= cnt) return;
    const int n0  = blockIdx.x * 128;  // over H

    __shared__ __align__(64) __bf16 sA[2][128][32];   // double-buffered async gather
    __shared__ __align__(64) FFN2U  sU;
    __shared__ int sAid[128];

    const int tid  = threadIdx.x;
    const int lane = tid & 31;
    const int wid  = tid >> 5;
    const int wm   = (wid >> 2) * 64;
    const int wn   = (wid & 3) * 32;

    const float* w2e   = w2 + (size_t)e * H_DIM * I_DIM;
    const int*   listE = lists + e * T_TOK + m0;

    v8f c[4][2];
#pragma unroll
    for (int mi = 0; mi < 4; ++mi)
#pragma unroll
        for (int ni = 0; ni < 2; ++ni) c[mi][ni] = (v8f){0, 0, 0, 0, 0, 0, 0, 0};

    const int r  = tid >> 3;
    const int c4 = (tid & 7) * 4;

    // pipelined B tile (VGPR-held) and double-buffered async A gather
    float4 pB[4];
    auto issueB = [&](int k0) {
#pragma unroll
        for (int it = 0; it < 4; ++it) {
            const int row = it * 32 + r;
            const size_t wb = (size_t)(n0 + row) * I_DIM + k0 + c4;
            pB[it] = *(const float4*)(w2e + wb);
            __builtin_prefetch(w2e + wb + 64, 0, 1);
        }
    };
    auto issueA = [&](int k0, int buf) {
#pragma unroll
        for (int it = 0; it < 2; ++it) {
            const int u   = it * 256 + tid;
            const int row = u >> 2;
            const int seg = u & 3;
            const int aid = (m0 + row < cnt) ? listE[row] : -1;
            const __bf16* g = (aid >= 0)
                ? (hmid + (size_t)aid * I_DIM + k0 + seg * 8) : zsrc;
            async_ld16(&sA[buf][row][seg * 8], g);
        }
    };
    // prologue: row ids + first async gather + first B tile
#pragma unroll
    for (int it = 0; it < 2; ++it) {
        const int u   = it * 256 + tid;
        const int row = u >> 2;
        if ((u & 3) == 0) sAid[row] = (m0 + row < cnt) ? listE[row] : -1;
    }
    issueA(0, 0);
    issueB(0);

    for (int k0 = 0; k0 < I_DIM; k0 += 32) {
        const int bi = (k0 >> 5) & 1;
        __syncthreads();   // previous step's fragment reads complete
#pragma unroll
        for (int it = 0; it < 4; ++it) {
            const int row = it * 32 + r;
            sU.b[row][c4 + 0] = f2bf(pB[it].x); sU.b[row][c4 + 1] = f2bf(pB[it].y);
            sU.b[row][c4 + 2] = f2bf(pB[it].z); sU.b[row][c4 + 3] = f2bf(pB[it].w);
        }
        wait_async0();     // current A buffer fully in LDS before the barrier
        __syncthreads();
        if (k0 + 32 < I_DIM) {          // overlap next tile with WMMA
            issueA(k0 + 32, bi ^ 1);
            issueB(k0 + 32);
        }

        const int khalf = (lane >> 4) * 16;
        const int mrow  = lane & 15;
        bf16x16 af[4], bf[2];
#pragma unroll
        for (int mi = 0; mi < 4; ++mi)
            af[mi] = *(const bf16x16*)&sA[bi][wm + mi * 16 + mrow][khalf];
#pragma unroll
        for (int ni = 0; ni < 2; ++ni)
            bf[ni] = *(const bf16x16*)&sU.b[wn + ni * 16 + mrow][khalf];
#pragma unroll
        for (int mi = 0; mi < 4; ++mi)
#pragma unroll
            for (int ni = 0; ni < 2; ++ni)
                c[mi][ni] = __builtin_amdgcn_wmma_f32_16x16x32_bf16(
                    false, af[mi], false, bf[ni], (short)0, c[mi][ni], false, false);
    }

    // epilogue: fp32 tile in LDS, coalesced async stores
    __syncthreads();
    const int nlane = lane & 15;
    const int mbase = (lane >> 4) * 8;
#pragma unroll
    for (int mi = 0; mi < 4; ++mi)
#pragma unroll
        for (int ni = 0; ni < 2; ++ni) {
            const int col = wn + ni * 16 + nlane;
#pragma unroll
            for (int j = 0; j < 8; ++j)
                sU.outF[wm + mi * 16 + mbase + j][col] = c[mi][ni][j];
        }
    __syncthreads();
#pragma unroll
    for (int it = 0; it < 16; ++it) {
        const int u   = it * 256 + tid;
        const int row = u >> 5;
        const int seg = u & 31;
        const int aid = sAid[row];
        float* g = (aid >= 0) ? (acc + (size_t)aid * H_DIM + n0 + seg * 4) : dump;
        async_st16(g, &sU.outF[row][seg * 4]);
    }
    wait_async0();
}

// ---------------------------------------------------------------- combine
__global__ __launch_bounds__(256) void moe_combine(
    const float* __restrict__ acc, const float* __restrict__ routew,
    float* __restrict__ out) {
    const int gid  = blockIdx.x * 256 + threadIdx.x;
    const int base = gid * 4;
    const int t = base / H_DIM;
    const int h = base % H_DIM;
    const float w0 = routew[t * 2 + 0];
    const float w1 = routew[t * 2 + 1];
    float4 a0 = *(const float4*)(acc + (size_t)(t * 2 + 0) * H_DIM + h);
    float4 a1 = *(const float4*)(acc + (size_t)(t * 2 + 1) * H_DIM + h);
    float4 o;
    o.x = w0 * a0.x + w1 * a1.x;
    o.y = w0 * a0.y + w1 * a1.y;
    o.z = w0 * a0.z + w1 * a1.z;
    o.w = w0 * a0.w + w1 * a1.w;
    *(float4*)(out + (size_t)t * H_DIM + h) = o;
}

// ---------------------------------------------------------------- launch
extern "C" void kernel_launch(void* const* d_in, const int* in_sizes, int n_in,
                              void* d_out, int out_size, void* d_ws, size_t ws_size,
                              hipStream_t stream) {
    const float* x    = (const float*)d_in[0];
    const float* gate = (const float*)d_in[1];
    const float* w1   = (const float*)d_in[2];
    const float* w3   = (const float*)d_in[3];
    const float* w2   = (const float*)d_in[4];
    float*       out  = (float*)d_out;
    char*        ws   = (char*)d_ws;

    // workspace layout
    int*    counts = (int*)(ws + 0);        // 8 ints
    float*  zsrc   = (float*)(ws + 256);    // 64B zero source for OOB gathers
    float*  dump   = (float*)(ws + 512);    // 512B discard sink for OOB scatters
    float*  routew = (float*)(ws + 4096);   // T*2 fp32 (64 KB)
    int*    lists  = (int*)(ws + 4096 + 65536);  // E*T int (256 KB)
    size_t  off    = 4096 + 65536 + (size_t)E_EXP * T_TOK * 4;
    off = (off + 4095) & ~(size_t)4095;
    __bf16* hmid = (__bf16*)(ws + off);     // 16384*7168 bf16 (~235 MB)
    off += (size_t)T_TOK * 2 * I_DIM * 2;
    off = (off + 4095) & ~(size_t)4095;
    float*  acc  = (float*)(ws + off);      // 16384*2048 fp32 (~134 MB)

    moe_zero<<<1, 32, 0, stream>>>(counts, zsrc);
    moe_router<<<T_TOK / 8, 256, 0, stream>>>(x, gate, routew, counts, lists);
    moe_ffn1<<<dim3(I_DIM / 128, T_TOK / 128, E_EXP), 256, 0, stream>>>(
        x, w1, w3, counts, lists, hmid, (__bf16*)dump);
    moe_ffn2<<<dim3(H_DIM / 128, T_TOK / 128, E_EXP), 256, 0, stream>>>(
        hmid, w2, counts, lists, acc, (const __bf16*)zsrc, dump);
    moe_combine<<<(T_TOK * H_DIM / 4) / 256, 256, 0, stream>>>(acc, routew, out);
}
